// MoBAAttention_10290741641659
// MI455X (gfx1250) — matrix operand order repaired and checked by reading, not compiled, not run on hardware
//
#include <hip/hip_runtime.h>
#include <hip/hip_bf16.h>
#include <math.h>

typedef __attribute__((ext_vector_type(16))) _Float16 v16h;
typedef __attribute__((ext_vector_type(8)))  _Float16 v8h;
typedef __attribute__((ext_vector_type(8)))  float    v8f;

#define H_SIZE   768
#define N_HEADS  12
#define HEAD_DIM 64
#define SEQ      2048
#define MBLOCK   256
#define NBLK     8

__device__ __forceinline__ v8f wmma16(v16h a, v16h b, v8f c) {
  // D = A(16x32 f16) * B(32x16 f16) + C(16x16 f32)
  return __builtin_amdgcn_wmma_f32_16x16x32_f16(false, a, false, b, (short)0, c,
                                                false, false);
}

// ---- tile loaders matching CDNA5 WMMA VGPR layouts (wave32) -----------------
// A (16xK slice of K=32): lane<16 -> K=[0..7,16..23]; lane>=16 -> K=[8..15,24..31]
__device__ __forceinline__ v16h ldA_h(const _Float16* base, int stride) {
  int l = threadIdx.x & 31;
  const _Float16* p = base + (size_t)(l & 15) * stride + ((l >> 4) << 3);
  v8h lo = *(const v8h*)p;
  v8h hi = *(const v8h*)(p + 16);
  v16h r;
#pragma unroll
  for (int i = 0; i < 8; ++i) { r[i] = lo[i]; r[8 + i] = hi[i]; }
  return r;
}
__device__ __forceinline__ v16h ldA_f32(const float* base, int stride) {
  int l = threadIdx.x & 31;
  const float4* p = (const float4*)(base + (size_t)(l & 15) * stride + ((l >> 4) << 3));
  float4 a = p[0], b = p[1], c = p[4], d = p[5];
  v16h r;
  r[0] = (_Float16)a.x;  r[1] = (_Float16)a.y;  r[2] = (_Float16)a.z;  r[3] = (_Float16)a.w;
  r[4] = (_Float16)b.x;  r[5] = (_Float16)b.y;  r[6] = (_Float16)b.z;  r[7] = (_Float16)b.w;
  r[8] = (_Float16)c.x;  r[9] = (_Float16)c.y;  r[10] = (_Float16)c.z; r[11] = (_Float16)c.w;
  r[12] = (_Float16)d.x; r[13] = (_Float16)d.y; r[14] = (_Float16)d.z; r[15] = (_Float16)d.w;
  return r;
}
// B (Kx16 slice of K=32): lane<16 -> K=[0..7,16..23]; lane>=16 -> K=[8..15,24..31]
__device__ __forceinline__ v16h ldB_h(const _Float16* base, int stride) {
  int l = threadIdx.x & 31;
  const _Float16* p = base + (size_t)(l & 15) * stride + ((l >> 4) << 3);
  v8h lo = *(const v8h*)p;
  v8h hi = *(const v8h*)(p + 16);
  v16h r;
#pragma unroll
  for (int i = 0; i < 8; ++i) { r[i] = lo[i]; r[8 + i] = hi[i]; }
  return r;
}
__device__ __forceinline__ v16h ldB_f32(const float* base, int stride) {
  int l = threadIdx.x & 31;
  const float4* p = (const float4*)(base + (size_t)(l & 15) * stride + ((l >> 4) << 3));
  v16h r;
#pragma unroll
  for (int j = 0; j < 4; ++j) {
    float4 q = p[j + (j & 2)];
    r[4 * j + 0] = (_Float16)q.x; r[4 * j + 1] = (_Float16)q.y;
    r[4 * j + 2] = (_Float16)q.z; r[4 * j + 3] = (_Float16)q.w;
  }
  return r;
}

// ---- K1: fused QKV projection (WMMA) + RoPE, writes f16 Q/K [h][s][64], V^T [h][64][s]
__global__ __launch_bounds__(128) void qkv_rope_kernel(
    const float* __restrict__ X, const float* __restrict__ Wq,
    const float* __restrict__ Wk, const float* __restrict__ Wv,
    _Float16* __restrict__ qh, _Float16* __restrict__ kh, _Float16* __restrict__ vT) {
  int st = blockIdx.x, h = blockIdx.y, mat = blockIdx.z;
  int wave = threadIdx.x >> 5, lane = threadIdx.x & 31;
  int s0 = st * 16;
  const float* W = (mat == 0) ? Wq : (mat == 1) ? Wk : Wv;
  int ob = h * HEAD_DIM + wave * 16;
  const float* Xb = X + (size_t)s0 * H_SIZE;
  const float* Wb = W + (size_t)ob * H_SIZE;

  // software-pipelined K loop: loads of tile t+1 issue before wmma of tile t
  v8f acc = {};
  v16h a = ldA_f32(Xb, H_SIZE);
  v16h b = ldB_f32(Wb, H_SIZE);
  const int NK = H_SIZE / 32;
#pragma unroll 1
  for (int t = 0; t < NK - 1; ++t) {
    v16h an = ldA_f32(Xb + (t + 1) * 32, H_SIZE);
    v16h bn = ldB_f32(Wb + (t + 1) * 32, H_SIZE);
    acc = wmma16(a, b, acc);
    a = an; b = bn;
  }
  acc = wmma16(a, b, acc);

  __shared__ float tile[16 * HEAD_DIM];
  int half = lane >> 4, col = lane & 15;
#pragma unroll
  for (int r = 0; r < 8; ++r)
    tile[(r + half * 8) * HEAD_DIM + wave * 16 + col] = acc[r];
  __syncthreads();

  for (int i = threadIdx.x; i < 16 * HEAD_DIM; i += 128) {
    int m = i >> 6, d = i & 63, s = s0 + m;
    float x = tile[i];
    if (mat < 2) {
      int j = d & 31;
      float invf = powf(10000.f, -(float)(2 * j) / (float)HEAD_DIM);
      float ang = (float)s * invf;
      float c = cosf(ang), sn = sinf(ang);
      float partner = (d < 32) ? -tile[m * HEAD_DIM + d + 32]
                               : tile[m * HEAD_DIM + d - 32];
      _Float16* dst = (mat == 0 ? qh : kh);
      dst[((size_t)h * SEQ + s) * HEAD_DIM + d] = (_Float16)(x * c + partner * sn);
    } else {
      vT[((size_t)h * HEAD_DIM + d) * SEQ + s] = (_Float16)x;
    }
  }
}

// ---- K2: per-block key means -------------------------------------------------
__global__ __launch_bounds__(64) void kmean_kernel(const _Float16* __restrict__ kh,
                                                   float* __restrict__ km) {
  int h = blockIdx.x, blk = blockIdx.y, d = threadIdx.x;
  const _Float16* p = kh + ((size_t)h * SEQ + blk * MBLOCK) * HEAD_DIM + d;
  float sum = 0.f;
  for (int j = 0; j < MBLOCK; ++j) sum += (float)p[(size_t)j * HEAD_DIM];
  km[((size_t)h * NBLK + blk) * HEAD_DIM + d] = sum * (1.f / MBLOCK);
}

// ---- K3a: affinity + top-3 (jax tie-breaking: desc value, lower index first) -
__global__ __launch_bounds__(256) void topk_kernel(
    const _Float16* __restrict__ qh, const float* __restrict__ km,
    int* __restrict__ idx, float* __restrict__ vals, int* __restrict__ hascur) {
  int h = blockIdx.y;
  int s = blockIdx.x * 256 + threadIdx.x;
  __shared__ float skm[NBLK * HEAD_DIM];
  for (int i = threadIdx.x; i < NBLK * HEAD_DIM; i += 256)
    skm[i] = km[(size_t)h * NBLK * HEAD_DIM + i];
  __syncthreads();

  const _Float16* qp = qh + ((size_t)h * SEQ + s) * HEAD_DIM;
  float dot[NBLK];
#pragma unroll
  for (int n = 0; n < NBLK; ++n) dot[n] = 0.f;
  for (int d = 0; d < HEAD_DIM; ++d) {
    float qv = (float)qp[d];
#pragma unroll
    for (int n = 0; n < NBLK; ++n) dot[n] += qv * skm[n * HEAD_DIM + d];
  }
  int cur = s >> 8;
  const float NEG = -__builtin_inff();
  float v0 = 0, v1 = 0, v2 = 0;
  int i0 = -1, i1 = -1, i2 = -1;
  for (int n = 0; n < NBLK; ++n) {
    float a = (n <= cur) ? dot[n] * 0.125f : NEG;
    if (i0 < 0)          { v0 = a; i0 = n; }
    else if (a > v0)     { v2 = v1; i2 = i1; v1 = v0; i1 = i0; v0 = a; i0 = n; }
    else if (i1 < 0)     { v1 = a; i1 = n; }
    else if (a > v1)     { v2 = v1; i2 = i1; v1 = a; i1 = n; }
    else if (i2 < 0)     { v2 = a; i2 = n; }
    else if (a > v2)     { v2 = a; i2 = n; }
  }
  size_t o = ((size_t)h * SEQ + s) * 3;
  idx[o] = i0; idx[o + 1] = i1; idx[o + 2] = i2;
  vals[o] = v0; vals[o + 1] = v1; vals[o + 2] = v2;
  hascur[(size_t)h * SEQ + s] = (i0 == cur) | (i1 == cur) | (i2 == cur);
}

// ---- K3b: cross-head "missing current block" reduction ----------------------
__global__ __launch_bounds__(256) void missing_kernel(const int* __restrict__ hascur,
                                                      int* __restrict__ missing) {
  int s = blockIdx.x * 256 + threadIdx.x;
  int all = 1;
#pragma unroll
  for (int h = 0; h < N_HEADS; ++h) all &= hascur[(size_t)h * SEQ + s];
  missing[s] = !all;
}

// ---- K3c: slot replacement + per-block log(count) ---------------------------
__global__ __launch_bounds__(256) void logc_kernel(
    const int* __restrict__ idx, const float* __restrict__ vals,
    const int* __restrict__ missing, float* __restrict__ logc) {
  int h = blockIdx.y;
  int s = blockIdx.x * 256 + threadIdx.x;
  size_t o = ((size_t)h * SEQ + s) * 3;
  int i0 = idx[o], i1 = idx[o + 1], i2 = idx[o + 2];
  float v0 = vals[o], v1 = vals[o + 1], v2 = vals[o + 2];
  if (missing[s]) {
    int ms = 0; float mv = v0;
    if (v1 < mv) { ms = 1; mv = v1; }
    if (v2 < mv) { ms = 2; }
    int cur = s >> 8;
    if (ms == 0) i0 = cur; else if (ms == 1) i1 = cur; else i2 = cur;
  }
  const float NEG = -__builtin_inff();
#pragma unroll
  for (int n = 0; n < NBLK; ++n) {
    int c = (i0 == n) + (i1 == n) + (i2 == n);
    logc[((size_t)h * SEQ + s) * NBLK + n] = (c > 0) ? logf((float)c) : NEG;
  }
}

// ---- K4: block-sparse flash attention (WMMA QK^T and PV) --------------------
__global__ __launch_bounds__(128) void attn_kernel(
    const _Float16* __restrict__ qh, const _Float16* __restrict__ kh,
    const _Float16* __restrict__ vT, const float* __restrict__ logc,
    float* __restrict__ ctx) {
  int h = blockIdx.x;
  int wave = threadIdx.x >> 5, lane = threadIdx.x & 31;
  int s0 = blockIdx.y * 64 + wave * 16;
  int half = lane >> 4, col = lane & 15;

  __shared__ __align__(16) _Float16 pbuf[4][16 * 32];
  _Float16* myp = pbuf[wave];

  const _Float16* qbase = qh + ((size_t)h * SEQ + s0) * HEAD_DIM;
  v16h qa0 = ldA_h(qbase, HEAD_DIM);
  v16h qa1 = ldA_h(qbase + 32, HEAD_DIM);

  v8f acc0 = {}, acc1 = {}, acc2 = {}, acc3 = {};
  const float NEG = -__builtin_inff();
  float mrun[8], lrun[8];
#pragma unroll
  for (int r = 0; r < 8; ++r) { mrun[r] = NEG; lrun[r] = 0.f; }

  for (int n = 0; n < NBLK; ++n) {
    float lc[8];
    bool anyf = false;
#pragma unroll
    for (int r = 0; r < 8; ++r) {
      int m = r + half * 8;
      lc[r] = logc[((size_t)h * SEQ + s0 + m) * NBLK + n];
      anyf |= (lc[r] > -1e30f);
    }
    if (!__any(anyf)) continue;  // whole 16-query tile skips this block

    for (int kt = 0; kt < MBLOCK / 32; ++kt) {
      int ks = n * MBLOCK + kt * 32;
      // S = Q(16x64) * K^T(64x32), two 16-key C tiles
      v8f sc0 = {}, sc1 = {};
      {
        const _Float16* kb = kh + ((size_t)h * SEQ + ks) * HEAD_DIM;
        sc0 = wmma16(qa0, ldB_h(kb, HEAD_DIM), sc0);
        sc0 = wmma16(qa1, ldB_h(kb + 32, HEAD_DIM), sc0);
        kb += 16 * HEAD_DIM;
        sc1 = wmma16(qa0, ldB_h(kb, HEAD_DIM), sc1);
        sc1 = wmma16(qa1, ldB_h(kb + 32, HEAD_DIM), sc1);
      }
      // issue V tiles now; consumed after the softmax + LDS transpose below
      const _Float16* vb = vT + (size_t)h * HEAD_DIM * SEQ + ks;
      v16h bv0 = ldB_h(vb + (size_t)0 * 16 * SEQ, SEQ);
      v16h bv1 = ldB_h(vb + (size_t)1 * 16 * SEQ, SEQ);
      v16h bv2 = ldB_h(vb + (size_t)2 * 16 * SEQ, SEQ);
      v16h bv3 = ldB_h(vb + (size_t)3 * 16 * SEQ, SEQ);

      float fac[8];
#pragma unroll
      for (int r = 0; r < 8; ++r) {
        float a0 = sc0[r] * 0.125f + lc[r];
        float a1 = sc1[r] * 0.125f + lc[r];
        float mt = fmaxf(a0, a1);
#pragma unroll
        for (int msk = 1; msk <= 8; msk <<= 1)
          mt = fmaxf(mt, __shfl_xor(mt, msk, 32));
        float mnew = fmaxf(mrun[r], mt);
        float p0, p1, f;
        if (mnew == NEG) { p0 = 0.f; p1 = 0.f; f = 1.f; }
        else {
          p0 = __expf(a0 - mnew);
          p1 = __expf(a1 - mnew);
          f = (mrun[r] == NEG) ? 0.f : __expf(mrun[r] - mnew);
        }
        float ls = p0 + p1;
#pragma unroll
        for (int msk = 1; msk <= 8; msk <<= 1)
          ls += __shfl_xor(ls, msk, 32);
        lrun[r] = lrun[r] * f + ls;
        mrun[r] = mnew;
        fac[r] = f;
        int m = r + half * 8;
        myp[m * 32 + col] = (_Float16)p0;        // C-layout -> row-major via LDS
        myp[m * 32 + 16 + col] = (_Float16)p1;
      }
#pragma unroll
      for (int r = 0; r < 8; ++r) {
        acc0[r] *= fac[r]; acc1[r] *= fac[r]; acc2[r] *= fac[r]; acc3[r] *= fac[r];
      }
      __builtin_amdgcn_wave_barrier();
      v16h pa;  // re-read P in A-matrix layout (DS ops in-order within wave)
      {
        const _Float16* p = myp + col * 32 + (half << 3);
        v8h lo = *(const v8h*)p;
        v8h hi = *(const v8h*)(p + 16);
#pragma unroll
        for (int i = 0; i < 8; ++i) { pa[i] = lo[i]; pa[8 + i] = hi[i]; }
      }
      __builtin_amdgcn_wave_barrier();
      acc0 = wmma16(pa, bv0, acc0);
      acc1 = wmma16(pa, bv1, acc1);
      acc2 = wmma16(pa, bv2, acc2);
      acc3 = wmma16(pa, bv3, acc3);
    }
  }
#pragma unroll
  for (int r = 0; r < 8; ++r) {
    float inv = (lrun[r] > 0.f) ? 1.f / lrun[r] : 0.f;
    int s = s0 + r + half * 8;
    float* o = ctx + (size_t)s * H_SIZE + h * HEAD_DIM;
    o[col]      = acc0[r] * inv;
    o[16 + col] = acc1[r] * inv;
    o[32 + col] = acc2[r] * inv;
    o[48 + col] = acc3[r] * inv;
  }
}

// ---- K5: output projection (WMMA) -------------------------------------------
__global__ __launch_bounds__(128) void oproj_kernel(const float* __restrict__ ctx,
                                                    const float* __restrict__ Wo,
                                                    float* __restrict__ out) {
  int st = blockIdx.x, ct = blockIdx.y;
  int wave = threadIdx.x >> 5, lane = threadIdx.x & 31;
  int s0 = st * 16;
  int ob = ct * 64 + wave * 16;
  const float* Cb = ctx + (size_t)s0 * H_SIZE;
  const float* Wb = Wo + (size_t)ob * H_SIZE;

  v8f acc = {};
  v16h a = ldA_f32(Cb, H_SIZE);
  v16h b = ldB_f32(Wb, H_SIZE);
  const int NK = H_SIZE / 32;
#pragma unroll 1
  for (int t = 0; t < NK - 1; ++t) {
    v16h an = ldA_f32(Cb + (t + 1) * 32, H_SIZE);
    v16h bn = ldB_f32(Wb + (t + 1) * 32, H_SIZE);
    acc = wmma16(a, b, acc);
    a = an; b = bn;
  }
  acc = wmma16(a, b, acc);

  int half = lane >> 4, col = lane & 15;
#pragma unroll
  for (int r = 0; r < 8; ++r)
    out[(size_t)(s0 + r + half * 8) * H_SIZE + ob + col] = acc[r];
}

extern "C" void kernel_launch(void* const* d_in, const int* in_sizes, int n_in,
                              void* d_out, int out_size, void* d_ws, size_t ws_size,
                              hipStream_t stream) {
  const float* X  = (const float*)d_in[0];
  const float* Wq = (const float*)d_in[1];
  const float* Wk = (const float*)d_in[2];
  const float* Wv = (const float*)d_in[3];
  const float* Wo = (const float*)d_in[4];
  float* out = (float*)d_out;
  (void)in_sizes; (void)n_in; (void)out_size; (void)ws_size;

  char* ws = (char*)d_ws;
  size_t off = 0;
  auto alloc = [&](size_t bytes) -> void* {
    void* p = ws + off;
    off = (off + bytes + 255) & ~(size_t)255;
    return p;
  };
  const size_t HE = (size_t)N_HEADS * SEQ * HEAD_DIM;
  _Float16* qh = (_Float16*)alloc(HE * 2);
  _Float16* kh = (_Float16*)alloc(HE * 2);
  _Float16* vT = (_Float16*)alloc(HE * 2);
  float* km     = (float*)alloc((size_t)N_HEADS * NBLK * HEAD_DIM * 4);
  int*   idx    = (int*)alloc((size_t)N_HEADS * SEQ * 3 * 4);
  float* vals   = (float*)alloc((size_t)N_HEADS * SEQ * 3 * 4);
  int*   hascur = (int*)alloc((size_t)N_HEADS * SEQ * 4);
  int*   missing= (int*)alloc((size_t)SEQ * 4);
  float* logc   = (float*)alloc((size_t)N_HEADS * SEQ * NBLK * 4);
  float* ctx    = (float*)alloc((size_t)SEQ * H_SIZE * 4);

  qkv_rope_kernel<<<dim3(SEQ / 16, N_HEADS, 3), 128, 0, stream>>>(X, Wq, Wk, Wv, qh, kh, vT);
  kmean_kernel<<<dim3(N_HEADS, NBLK), 64, 0, stream>>>(kh, km);
  topk_kernel<<<dim3(SEQ / 256, N_HEADS), 256, 0, stream>>>(qh, km, idx, vals, hascur);
  missing_kernel<<<dim3(SEQ / 256), 256, 0, stream>>>(hascur, missing);
  logc_kernel<<<dim3(SEQ / 256, N_HEADS), 256, 0, stream>>>(idx, vals, missing, logc);
  attn_kernel<<<dim3(N_HEADS, SEQ / 64), 128, 0, stream>>>(qh, kh, vT, logc, ctx);
  oproj_kernel<<<dim3(SEQ / 16, H_SIZE / 64), 128, 0, stream>>>(ctx, Wo, out);
}
